// HypER_19885698580525
// MI455X (gfx1250) — matrix operand order repaired and checked by reading, not compiled, not run on hardware
//
#include <hip/hip_runtime.h>
#include <hip/hip_bf16.h>
#include <math.h>

// ---------------- problem constants ----------------
#define NF     32
#define LF     9
#define D_E    200
#define D_R    200
#define L_OUT  192            // D_E - LF + 1
#define NENT   100000
#define BATCH  1024
#define FC1LEN (NF*LF)        // 288
#define FCLEN  (NF*L_OUT)     // 6144
#define KPAD   224            // D_E padded up to multiple of 32
#define NPAD   208            // D_E padded up to multiple of 16 (fc_w rows)
#define EPSF   1e-5f

typedef _Float16 h16 __attribute__((ext_vector_type(16)));
typedef _Float16 h8  __attribute__((ext_vector_type(8)));
typedef float    f8  __attribute__((ext_vector_type(8)));

// ---------------- ws layout (floats / halves) ----------------
// stats:   [0]   bn0_sum, [1] bn0_sq
//          [8..39]   bn1_sum[f], [40..71] bn1_sq[f]
//          [80..279] bn2_sum[d], [280..479] bn2_sq[d]
#define STATS_FLOATS 512
#define K_ELEMS      (BATCH*FC1LEN)       // 294912
#define CONV_ELEMS   (BATCH*FCLEN)        // 6291456
#define Y_ELEMS      (BATCH*D_E)          // 204800
#define CONVH_ELEMS  (BATCH*FCLEN)        // 6291456 halves
#define FCWH_ELEMS   (NPAD*FCLEN)         // 1277952 halves
#define YH_ELEMS     (BATCH*KPAD)         // 229376 halves

// ---------------- WMMA helpers ----------------
__device__ __forceinline__ f8 wmma16x16x32(h16 a, h16 b, f8 c) {
    return __builtin_amdgcn_wmma_f32_16x16x32_f16(
        /*neg_a=*/false, a, /*neg_b=*/false, b,
        /*c_mod=*/(short)0, c, /*reuse_a=*/false, /*reuse_b=*/false);
}

// A fragment: 16x32 f16, row-major source with leading dim `ld` (halves).
// base already points at (row0, k0).
__device__ __forceinline__ h16 load_a_frag(const _Float16* __restrict__ base,
                                           int ld, int lane) {
    int m  = lane & 15;
    int hi = lane >> 4;
    const _Float16* p = base + (size_t)m * ld + hi * 8;
    h8 lo = *(const h8*)(p);        // K = hi*8 .. hi*8+7
    h8 hh = *(const h8*)(p + 16);   // K = 16+hi*8 .. 16+hi*8+7
    return __builtin_shufflevector(lo, hh,
        0,1,2,3,4,5,6,7,8,9,10,11,12,13,14,15);
}

// B fragment: 32x16 f16 where B[k][n] = W[n][k]; W row-major, leading dim ld.
// base points at (col0 row of W, k0).
__device__ __forceinline__ h16 load_b_frag(const _Float16* __restrict__ base,
                                           int ld, int lane) {
    int n  = lane & 15;
    int hi = lane >> 4;
    const _Float16* p = base + (size_t)n * ld + hi * 8;
    h8 lo = *(const h8*)(p);
    h8 hh = *(const h8*)(p + 16);
    return __builtin_shufflevector(lo, hh,
        0,1,2,3,4,5,6,7,8,9,10,11,12,13,14,15);
}

// ---------------- small kernels ----------------
__global__ void zero_stats_kernel(float* __restrict__ stats) {
    int i = blockIdx.x * 256 + threadIdx.x;
    if (i < STATS_FLOATS) stats[i] = 0.0f;
}

__device__ __forceinline__ void reduce2_atomic(float a, float q,
                                               float* dst_a, float* dst_q) {
    for (int o = 16; o > 0; o >>= 1) {
        a += __shfl_down(a, o);
        q += __shfl_down(q, o);
    }
    __shared__ float sa[8], sq[8];
    int t = threadIdx.x, lane = t & 31, w = t >> 5;
    if (lane == 0) { sa[w] = a; sq[w] = q; }
    __syncthreads();
    if (t == 0) {
        float A = 0.f, Q = 0.f;
        for (int i = 0; i < 8; ++i) { A += sa[i]; Q += sq[i]; }
        atomicAdd(dst_a, A);
        atomicAdd(dst_q, Q);
    }
}

// BN0 statistics over all B*D_E elements of e1. grid=200, block=256.
__global__ void bn0_reduce_kernel(const float* __restrict__ e1,
                                  float* __restrict__ stats) {
    int t = threadIdx.x;
    float a = 0.f, q = 0.f;
    size_t base = (size_t)blockIdx.x * 1024;
    for (int j = 0; j < 4; ++j) {
        float v = e1[base + j * 256 + t];
        a += v; q += v * v;
    }
    reduce2_atomic(a, q, &stats[0], &stats[1]);
}

// k = r @ fc1_w.T + fc1_b. grid=1152, block=256.
__global__ void fc1_kernel(const float* __restrict__ r,
                           const float* __restrict__ w,
                           const float* __restrict__ bias,
                           float* __restrict__ k) {
    int idx = blockIdx.x * 256 + threadIdx.x;
    if (idx >= BATCH * FC1LEN) return;
    int b = idx / FC1LEN, f = idx % FC1LEN;
    const float* rr = r + (size_t)b * D_R;
    const float* ww = w + (size_t)f * D_R;
    float acc = bias[f];
    for (int j = 0; j < D_R; ++j) acc = fmaf(rr[j], ww[j], acc);
    k[idx] = acc;
}

// conv[b,f,l] = sum_j bn0(e1[b,l+j]) * k[b,f,j]. grid=B*NF, block=192.
__global__ void conv_kernel(const float* __restrict__ e1,
                            const float* __restrict__ k,
                            const float* __restrict__ stats,
                            const float* __restrict__ g0,
                            const float* __restrict__ b0,
                            float* __restrict__ conv) {
    int bf = blockIdx.x;
    int b = bf >> 5, f = bf & 31;
    __shared__ float ks[LF];
    __shared__ float xs[D_E];
    int l = threadIdx.x;
    const float inv_n = 1.0f / (float)(BATCH * D_E);
    float mean = stats[0] * inv_n;
    float var  = stats[1] * inv_n - mean * mean;
    float rs   = rsqrtf(var + EPSF);
    float sg   = rs * g0[0];
    float sb   = b0[0] - mean * sg;
    if (l < LF) ks[l] = k[(size_t)b * FC1LEN + f * LF + l];
    for (int i = l; i < D_E; i += 192)
        xs[i] = fmaf(e1[(size_t)b * D_E + i], sg, sb);
    __syncthreads();
    float acc = 0.f;
    #pragma unroll
    for (int j = 0; j < LF; ++j) acc = fmaf(xs[l + j], ks[j], acc);
    conv[(size_t)b * FCLEN + f * L_OUT + l] = acc;
}

// BN1 statistics per filter f over B*L values. grid=(64,32), block=256.
__global__ void bn1_reduce_kernel(const float* __restrict__ conv,
                                  float* __restrict__ stats) {
    int f = blockIdx.y;
    int t = threadIdx.x;
    float a = 0.f, q = 0.f;
    for (int i = 0; i < 12; ++i) {
        int j = blockIdx.x * 3072 + i * 256 + t;   // 0..196607
        int b = j / L_OUT, l = j % L_OUT;
        float v = conv[(size_t)b * FCLEN + f * L_OUT + l];
        a += v; q += v * v;
    }
    reduce2_atomic(a, q, &stats[8 + f], &stats[40 + f]);
}

// Apply BN1 + convert conv -> f16. grid=24576, block=256.
__global__ void convh_kernel(const float* __restrict__ conv,
                             const float* __restrict__ stats,
                             const float* __restrict__ g1,
                             const float* __restrict__ b1,
                             _Float16* __restrict__ convh) {
    int idx = blockIdx.x * 256 + threadIdx.x;
    int f = (idx / L_OUT) & 31;
    const float inv_n = 1.0f / (float)(BATCH * L_OUT);
    float m  = stats[8 + f] * inv_n;
    float v  = stats[40 + f] * inv_n - m * m;
    float rs = rsqrtf(v + EPSF);
    float val = (conv[idx] - m) * rs * g1[f] + b1[f];
    convh[idx] = (_Float16)val;
}

// fc_w (200x6144) -> f16, rows padded to 208 with zeros. grid=4992, block=256.
__global__ void fcwh_kernel(const float* __restrict__ fcw,
                            _Float16* __restrict__ fcwh) {
    int idx = blockIdx.x * 256 + threadIdx.x;        // < NPAD*FCLEN
    int d = idx / FCLEN;
    float v = (d < D_E) ? fcw[idx] : 0.0f;
    fcwh[idx] = (_Float16)v;
}

// y = convh @ fcwh^T + fc_b via WMMA. grid=(13,8), block=256 (8 waves).
__global__ void mid_gemm_kernel(const _Float16* __restrict__ convh,
                                const _Float16* __restrict__ fcwh,
                                const float* __restrict__ fcb,
                                float* __restrict__ y) {
    int t = threadIdx.x, wave = t >> 5, lane = t & 31;
    int d0   = blockIdx.x * 16;
    int row0 = blockIdx.y * 128 + wave * 16;
    const _Float16* abase = convh + (size_t)row0 * FCLEN;
    const _Float16* bbase = fcwh  + (size_t)d0   * FCLEN;
    f8 acc = {};
    for (int k0 = 0; k0 < FCLEN; k0 += 32) {
        h16 a = load_a_frag(abase + k0, FCLEN, lane);
        h16 b = load_b_frag(bbase + k0, FCLEN, lane);
        acc = wmma16x16x32(a, b, acc);
    }
    int n = lane & 15, hi = lane >> 4;
    int d = d0 + n;
    if (d < D_E) {
        float bb = fcb[d];
        #pragma unroll
        for (int r = 0; r < 8; ++r) {
            int m = r + hi * 8;
            y[(size_t)(row0 + m) * D_E + d] = acc[r] + bb;
        }
    }
}

// BN2 statistics per feature d over batch. grid=200, block=256.
__global__ void bn2_reduce_kernel(const float* __restrict__ y,
                                  float* __restrict__ stats) {
    int d = blockIdx.x, t = threadIdx.x;
    float a = 0.f, q = 0.f;
    for (int i = 0; i < 4; ++i) {
        float v = y[(size_t)(i * 256 + t) * D_E + d];
        a += v; q += v * v;
    }
    for (int o = 16; o > 0; o >>= 1) {
        a += __shfl_down(a, o);
        q += __shfl_down(q, o);
    }
    __shared__ float sa[8], sq[8];
    int lane = t & 31, w = t >> 5;
    if (lane == 0) { sa[w] = a; sq[w] = q; }
    __syncthreads();
    if (t == 0) {
        float A = 0.f, Q = 0.f;
        for (int i = 0; i < 8; ++i) { A += sa[i]; Q += sq[i]; }
        stats[80 + d]  = A;
        stats[280 + d] = Q;
    }
}

// Apply BN2 + ReLU, convert y -> f16 padded to KPAD. grid=896, block=256.
__global__ void yh_kernel(const float* __restrict__ y,
                          const float* __restrict__ stats,
                          const float* __restrict__ g2,
                          const float* __restrict__ b2,
                          _Float16* __restrict__ yh) {
    int idx = blockIdx.x * 256 + threadIdx.x;        // < BATCH*KPAD
    int b = idx / KPAD, d = idx % KPAD;
    float v = 0.f;
    if (d < D_E) {
        float m  = stats[80 + d]  * (1.0f / BATCH);
        float va = stats[280 + d] * (1.0f / BATCH) - m * m;
        float val = (y[(size_t)b * D_E + d] - m) * rsqrtf(va + EPSF) * g2[d] + b2[d];
        v = fmaxf(val, 0.f);
    }
    yh[idx] = (_Float16)v;
}

// scores = sigmoid(yh @ E^T + b) via WMMA. grid=(6250,8), block=256 (8 waves).
// Each block: 16 entities (E tile converted f32->f16 into LDS, shared by
// 8 waves covering 128 batch rows). K = 200 padded to 224 -> 7 WMMA steps.
__global__ void score_gemm_kernel(const _Float16* __restrict__ yh,
                                  const float* __restrict__ E,
                                  const float* __restrict__ bias,
                                  float* __restrict__ out) {
    __shared__ __align__(32) _Float16 Esh[16 * KPAD];
    __shared__ float bsh[16];
    int t = threadIdx.x;
    int ent0 = blockIdx.x * 16;
    int b0   = blockIdx.y * 128;
    for (int i = t; i < 16 * KPAD; i += 256) {
        int rowe = i / KPAD, c = i % KPAD;
        float v = (c < D_E) ? E[(size_t)(ent0 + rowe) * D_E + c] : 0.0f;
        Esh[i] = (_Float16)v;
    }
    if (t < 16) bsh[t] = bias[ent0 + t];
    __syncthreads();

    int wave = t >> 5, lane = t & 31;
    int row0 = b0 + wave * 16;
    const _Float16* abase = yh + (size_t)row0 * KPAD;
    f8 acc = {};
    #pragma unroll
    for (int kk = 0; kk < KPAD / 32; ++kk) {
        h16 a = load_a_frag(abase + kk * 32, KPAD, lane);
        h16 b = load_b_frag(Esh + kk * 32, KPAD, lane);
        acc = wmma16x16x32(a, b, acc);
    }
    int n = lane & 15, hi = lane >> 4;
    float bb = bsh[n];
    #pragma unroll
    for (int r = 0; r < 8; ++r) {
        int m = r + hi * 8;
        float v = acc[r] + bb;
        v = 1.0f / (1.0f + __expf(-v));
        out[(size_t)(row0 + m) * NENT + ent0 + n] = v;
    }
}

// ---------------- launch ----------------
extern "C" void kernel_launch(void* const* d_in, const int* in_sizes, int n_in,
                              void* d_out, int out_size, void* d_ws, size_t ws_size,
                              hipStream_t stream) {
    const float* e1    = (const float*)d_in[0];
    const float* r     = (const float*)d_in[1];
    const float* fc1_w = (const float*)d_in[2];
    const float* fc1_b = (const float*)d_in[3];
    const float* fc_w  = (const float*)d_in[4];
    const float* fc_b  = (const float*)d_in[5];
    const float* E     = (const float*)d_in[6];
    const float* bvec  = (const float*)d_in[7];
    const float* bn0_g = (const float*)d_in[8];
    const float* bn0_b = (const float*)d_in[9];
    const float* bn1_g = (const float*)d_in[10];
    const float* bn1_b = (const float*)d_in[11];
    const float* bn2_g = (const float*)d_in[12];
    const float* bn2_b = (const float*)d_in[13];
    float* out = (float*)d_out;

    float* wsf   = (float*)d_ws;
    float* stats = wsf;
    float* kbuf  = wsf + STATS_FLOATS;
    float* conv  = kbuf + K_ELEMS;
    float* ybuf  = conv + CONV_ELEMS;
    _Float16* convh = (_Float16*)(ybuf + Y_ELEMS);
    _Float16* fcwh  = convh + CONVH_ELEMS;
    _Float16* yh    = fcwh + FCWH_ELEMS;

    zero_stats_kernel<<<2, 256, 0, stream>>>(stats);
    bn0_reduce_kernel<<<200, 256, 0, stream>>>(e1, stats);
    fc1_kernel<<<(BATCH * FC1LEN + 255) / 256, 256, 0, stream>>>(r, fc1_w, fc1_b, kbuf);
    conv_kernel<<<BATCH * NF, 192, 0, stream>>>(e1, kbuf, stats, bn0_g, bn0_b, conv);
    bn1_reduce_kernel<<<dim3(64, 32), 256, 0, stream>>>(conv, stats);
    convh_kernel<<<CONV_ELEMS / 256, 256, 0, stream>>>(conv, stats, bn1_g, bn1_b, convh);
    fcwh_kernel<<<FCWH_ELEMS / 256, 256, 0, stream>>>(fc_w, fcwh);
    mid_gemm_kernel<<<dim3(13, 8), 256, 0, stream>>>(convh, fcwh, fc_b, ybuf);
    bn2_reduce_kernel<<<200, 256, 0, stream>>>(ybuf, stats);
    yh_kernel<<<YH_ELEMS / 256, 256, 0, stream>>>(ybuf, stats, bn2_g, bn2_b, yh);
    score_gemm_kernel<<<dim3(NENT / 16, BATCH / 128), 256, 0, stream>>>(yh, E, bvec, out);
}
